// _SelfAttentionBlock_71210557767854
// MI455X (gfx1250) — matrix operand order repaired and checked by reading, not compiled, not run on hardware
//
#include <hip/hip_runtime.h>
#include <hip/hip_bf16.h>

// ---------------------------------------------------------------------------
// Fixed problem sizes from the reference
//   x:[8,512,64,64]  Wk:[256,512] Wv:[256,512] Ww:[512,256]  N=H*W=4096
// d_out = concat(out[8,512,4096], feat[8,256,4096], feat, value[8,256,4096])
// ---------------------------------------------------------------------------
#define BB   8
#define CC   512
#define CK   256
#define CV   256
#define CO   512
#define NN   4096
#define EPSF 1e-5f

typedef unsigned short u16;
typedef __attribute__((ext_vector_type(8)))  unsigned short u16x8;
typedef __attribute__((ext_vector_type(16))) unsigned short u16x16;
typedef __attribute__((ext_vector_type(16))) __bf16         bf16x16;
typedef __attribute__((ext_vector_type(8)))  float          f32x8;

__device__ __forceinline__ u16 f2bf(float f) {
    unsigned u = __builtin_bit_cast(unsigned, f);
    unsigned r = u + 0x7FFFu + ((u >> 16) & 1u);   // round-to-nearest-even
    return (u16)(r >> 16);
}

__device__ __forceinline__ bf16x16 mkfrag(u16x8 lo, u16x8 hi) {
    u16x16 v = __builtin_shufflevector(lo, hi, 0,1,2,3,4,5,6,7,8,9,10,11,12,13,14,15);
    return __builtin_bit_cast(bf16x16, v);
}

__device__ __forceinline__ f32x8 wmma_bf16(bf16x16 a, bf16x16 b, f32x8 c) {
    return __builtin_amdgcn_wmma_f32_16x16x32_bf16(
        /*neg_a=*/false, a, /*neg_b=*/false, b,
        /*c_mod=*/(short)0, c, /*reuse_a=*/false, /*reuse_b=*/false);
}

// ---------------------------------------------------------------------------
// f32 -> bf16 elementwise convert (weights)
// ---------------------------------------------------------------------------
__global__ void k_cvt_bf16(const float* __restrict__ src, u16* __restrict__ dst, int n) {
    int i = blockIdx.x * blockDim.x + threadIdx.x;
    if (i < n) dst[i] = f2bf(src[i]);
}

// ---------------------------------------------------------------------------
// LDS-tiled transpose + convert:  x f32 [B][C][N]  ->  xt bf16 [B][N][C]
// block (32,8), grid (N/32, C/32, B)
// ---------------------------------------------------------------------------
__global__ void k_transpose_cvt(const float* __restrict__ x, u16* __restrict__ xt) {
    __shared__ float tile[32][33];
    int b  = blockIdx.z;
    int n0 = blockIdx.x * 32, c0 = blockIdx.y * 32;
    int tx = threadIdx.x, ty = threadIdx.y;
    for (int i = 0; i < 4; ++i) {
        int c = c0 + ty + i * 8;
        tile[ty + i * 8][tx] = x[((size_t)b * CC + c) * NN + n0 + tx];
    }
    __syncthreads();
    for (int i = 0; i < 4; ++i) {
        int n = n0 + ty + i * 8;
        xt[((size_t)b * NN + n) * CC + c0 + tx] = f2bf(tile[tx][ty + i * 8]);
    }
}

// ---------------------------------------------------------------------------
// WMMA GEMM:  out[b][m][n] = sum_k W[m][k] * Xt[b][n][k] + bias[m]
//   W  : bf16 [M][K] row-major            (A fragments: contiguous 16B loads)
//   Xt : bf16 [B][N][K] (K contiguous)    (B fragments: contiguous 16B loads)
// One wave per 16x64 strip: A fragment reused across 4 N-tiles (4 accums).
// grid (N/64, M/16, B), block 32.
// ---------------------------------------------------------------------------
__global__ void k_gemm(const u16* __restrict__ Wb, const u16* __restrict__ Xt,
                       const float* __restrict__ bias,
                       float* __restrict__ outF, u16* __restrict__ outB,
                       int M, int K, int Nn) {
    int lane = threadIdx.x;
    int h = lane >> 4, cl = lane & 15;
    int n0 = blockIdx.x * 64, m0 = blockIdx.y * 16, b = blockIdx.z;

    const u16* ap = Wb + (size_t)(m0 + cl) * K + h * 16;           // A: row m, 16 contig K
    const u16* bp0 = Xt + ((size_t)b * Nn + n0 + cl) * K + h * 16; // B: col n, 16 contig K
    const u16* bp1 = bp0 + (size_t)16 * K;
    const u16* bp2 = bp0 + (size_t)32 * K;
    const u16* bp3 = bp0 + (size_t)48 * K;

    f32x8 acc0 = {}, acc1 = {}, acc2 = {}, acc3 = {};
    for (int c0 = 0; c0 < K; c0 += 32) {
        __builtin_prefetch(bp0 + c0 + 128, 0, 1);
        __builtin_prefetch(bp1 + c0 + 128, 0, 1);
        __builtin_prefetch(bp2 + c0 + 128, 0, 1);
        __builtin_prefetch(bp3 + c0 + 128, 0, 1);
        bf16x16 av = mkfrag(*(const u16x8*)(ap + c0), *(const u16x8*)(ap + c0 + 8));
        acc0 = wmma_bf16(av, mkfrag(*(const u16x8*)(bp0 + c0), *(const u16x8*)(bp0 + c0 + 8)), acc0);
        acc1 = wmma_bf16(av, mkfrag(*(const u16x8*)(bp1 + c0), *(const u16x8*)(bp1 + c0 + 8)), acc1);
        acc2 = wmma_bf16(av, mkfrag(*(const u16x8*)(bp2 + c0), *(const u16x8*)(bp2 + c0 + 8)), acc2);
        acc3 = wmma_bf16(av, mkfrag(*(const u16x8*)(bp3 + c0), *(const u16x8*)(bp3 + c0 + 8)), acc3);
    }
    // D layout: lane l holds rows r+8h, col cl
    f32x8 accs[4] = {acc0, acc1, acc2, acc3};
    for (int t = 0; t < 4; ++t)
        for (int r = 0; r < 8; ++r) {
            int m = m0 + r + 8 * h, n = n0 + t * 16 + cl;
            float v = accs[t][r] + bias[m];
            size_t idx = ((size_t)b * M + m) * Nn + n;
            if (outF) outF[idx] = v;
            if (outB) outB[idx] = f2bf(v);
        }
}

// ---------------------------------------------------------------------------
// BatchNorm training stats per channel -> folded scale/shift
// grid 256 (one block per channel), block 256
// ---------------------------------------------------------------------------
__global__ void k_bnstats(const float* __restrict__ qk,
                          const float* __restrict__ gamma,
                          const float* __restrict__ beta,
                          float* __restrict__ scsh) {
    int ch = blockIdx.x, tid = threadIdx.x;
    float s = 0.f, s2 = 0.f;
    for (int b = 0; b < BB; ++b) {
        const float* p = qk + ((size_t)b * CK + ch) * NN;
        for (int i = tid; i < NN; i += 256) { float v = p[i]; s += v; s2 += v * v; }
    }
    __shared__ float rs[256], rq[256];
    rs[tid] = s; rq[tid] = s2;
    __syncthreads();
    for (int off = 128; off > 0; off >>= 1) {
        if (tid < off) { rs[tid] += rs[tid + off]; rq[tid] += rq[tid + off]; }
        __syncthreads();
    }
    if (tid == 0) {
        const float inv = 1.f / (float)(BB * NN);
        float mean = rs[0] * inv;
        float var  = rq[0] * inv - mean * mean;
        float sc   = gamma[ch] * rsqrtf(var + EPSF);
        scsh[ch]        = sc;
        scsh[CK + ch]   = beta[ch] - mean * sc;
    }
}

// ---------------------------------------------------------------------------
// BN+ReLU epilogue: feat = relu(qk*sc+sh)
//   writes feat f32 to both tuple slots, and bf16 transposed [B][N][Ck]
// block (32,8), grid (N/32, Ck/32, B)
// ---------------------------------------------------------------------------
__global__ void k_feat(const float* __restrict__ qk, const float* __restrict__ scsh,
                       float* __restrict__ f1, float* __restrict__ f2,
                       u16* __restrict__ featT) {
    __shared__ float tile[32][33];
    int b  = blockIdx.z;
    int n0 = blockIdx.x * 32, c0 = blockIdx.y * 32;
    int tx = threadIdx.x, ty = threadIdx.y;
    for (int i = 0; i < 4; ++i) {
        int c = c0 + ty + i * 8;
        size_t o = ((size_t)b * CK + c) * NN + n0 + tx;
        float v = fmaxf(qk[o] * scsh[c] + scsh[CK + c], 0.f);
        f1[o] = v; f2[o] = v;
        tile[ty + i * 8][tx] = v;
    }
    __syncthreads();
    for (int i = 0; i < 4; ++i) {
        int n = n0 + ty + i * 8;
        featT[((size_t)b * NN + n) * CK + c0 + tx] = f2bf(tile[tx][ty + i * 8]);
    }
}

// ---------------------------------------------------------------------------
// Async staging of one 32-key x 256-channel V block (16KB) into LDS.
// Layout in LDS: row cv (256 rows) of 32 keys (64B), row stride 64B.
// Each of the 64 threads issues 16 b128 async loads (ASYNCcnt-tracked).
// ---------------------------------------------------------------------------
__device__ __forceinline__ void stage_v_async(const u16* __restrict__ Vb,
                                              u16* __restrict__ dst,
                                              int j, int tid) {
#pragma unroll
    for (int it = 0; it < 16; ++it) {
        int q  = it * 64 + tid;           // 1024 chunks of 16B
        int cv = q >> 2, kp = q & 3;
        const u16* g = Vb + (size_t)cv * NN + j + kp * 8;
        unsigned lds = (unsigned)(uintptr_t)(dst + q * 8);
        unsigned long long ga = (unsigned long long)(uintptr_t)g;
        asm volatile("global_load_async_to_lds_b128 %0, %1, off"
                     :: "v"(lds), "v"(ga) : "memory");
    }
}

// ---------------------------------------------------------------------------
// Fused flash attention over N=4096 keys per batch.
//   featT : bf16 [B][N][Ck]   (Q rows as A-frags, K cols as B-frags)
//   Vbf   : bf16 [B][Cv][N]   (async-staged to LDS, shared by both waves)
//   ctxT  : bf16 [B][N][Cv]   (output, B-frag layout for final projection)
// 2 waves/block, each wave owns 16 query rows; Q resident (8 frags),
// 16 f32x8 accumulators (16x256 ctx tile), online softmax via xor shuffles,
// P converted D-layout -> A-layout through LDS, V double-buffered via
// global_load_async_to_lds_b128 + s_wait_asynccnt.
// ---------------------------------------------------------------------------
__global__ void k_flash(const u16* __restrict__ featT, const u16* __restrict__ Vbf,
                        u16* __restrict__ ctxT) {
    __shared__ __align__(16) u16 pbuf[2 * 16 * 32];   // P staging, per wave 1KB
    __shared__ __align__(16) u16 vbuf[2][8192];       // V double buffer, 16KB each
    const float scale = 0.0625f;  // Ck^-0.5 = 1/16

    int tid  = threadIdx.x;
    int lane = tid & 31, w = tid >> 5;
    int h = lane >> 4, cl = lane & 15;
    int b  = blockIdx.y;
    int i0 = (blockIdx.x * 2 + w) * 16;   // first query row of this wave
    const u16* Vb = Vbf + (size_t)b * CV * NN;

    // resident Q fragments (16 rows x 256 channels)
    bf16x16 qf[8];
    {
        const u16* qb = featT + ((size_t)b * NN + i0 + cl) * CK + h * 16;
        for (int kc = 0; kc < 8; ++kc) {
            u16x8 lo = *(const u16x8*)(qb + kc * 32);
            u16x8 hi = *(const u16x8*)(qb + kc * 32 + 8);
            qf[kc] = mkfrag(lo, hi);
        }
    }

    f32x8 acc[16];
    for (int t = 0; t < 16; ++t) acc[t] = (f32x8){};
    float mrun[8], lrun[8];
    for (int r = 0; r < 8; ++r) { mrun[r] = -3.0e38f; lrun[r] = 0.f; }

    u16* pl = pbuf + w * (16 * 32);

    // preload V block for j=0 into buffer 0
    stage_v_async(Vb, vbuf[0], 0, tid);

    for (int j = 0; j < NN; j += 32) {
        int cur = (j >> 5) & 1;

        // issue async stage of the next V block into the other buffer
        if (j + 32 < NN) stage_v_async(Vb, vbuf[cur ^ 1], j + 32, tid);

        // --- S = scale * Q . K^T for two 16-key tiles (global, L2-resident) ---
        f32x8 s0 = {}, s1 = {};
        const u16* kb0 = featT + ((size_t)b * NN + j + cl) * CK + h * 16;
        const u16* kb1 = kb0 + (size_t)16 * CK;
        for (int kc = 0; kc < 8; ++kc) {
            u16x8 x0 = *(const u16x8*)(kb0 + kc * 32);
            u16x8 x1 = *(const u16x8*)(kb0 + kc * 32 + 8);
            s0 = wmma_bf16(qf[kc], mkfrag(x0, x1), s0);
            u16x8 y0 = *(const u16x8*)(kb1 + kc * 32);
            u16x8 y1 = *(const u16x8*)(kb1 + kc * 32 + 8);
            s1 = wmma_bf16(qf[kc], mkfrag(y0, y1), s1);
        }

        // --- online softmax (rows r+8h live in 16-lane halves) ---
        float alpha[8];
        for (int r = 0; r < 8; ++r) {
            float a = s0[r] * scale, c = s1[r] * scale;
            s0[r] = a; s1[r] = c;
            float rm = fmaxf(a, c);
            rm = fmaxf(rm, __shfl_xor(rm, 1, 32));
            rm = fmaxf(rm, __shfl_xor(rm, 2, 32));
            rm = fmaxf(rm, __shfl_xor(rm, 4, 32));
            rm = fmaxf(rm, __shfl_xor(rm, 8, 32));
            float mn = fmaxf(mrun[r], rm);
            alpha[r] = __expf(mrun[r] - mn);
            mrun[r] = mn;
        }
        for (int r = 0; r < 8; ++r) {
            float p0 = __expf(s0[r] - mrun[r]);
            float p1 = __expf(s1[r] - mrun[r]);
            s0[r] = p0; s1[r] = p1;
            float ps = p0 + p1;
            ps += __shfl_xor(ps, 1, 32);
            ps += __shfl_xor(ps, 2, 32);
            ps += __shfl_xor(ps, 4, 32);
            ps += __shfl_xor(ps, 8, 32);
            lrun[r] = lrun[r] * alpha[r] + ps;
        }
        for (int t = 0; t < 16; ++t)
            for (int r = 0; r < 8; ++r) acc[t][r] *= alpha[r];

        // --- P: D-layout -> A-layout via LDS (per-wave private 1KB) ---
        for (int r = 0; r < 8; ++r) {
            int row = r + 8 * h;
            pl[row * 32 + cl]      = f2bf(s0[r]);
            pl[row * 32 + 16 + cl] = f2bf(s1[r]);
        }
        asm volatile("s_wait_dscnt 0" ::: "memory");
        const u16* pr = pl + cl * 32 + h * 16;
        bf16x16 pf = mkfrag(*(const u16x8*)pr, *(const u16x8*)(pr + 8));

        // --- wait for current V buffer (keep the 16 newest async loads in
        //     flight), then sync both waves before consuming ---
        if (j + 32 < NN) asm volatile("s_wait_asynccnt 0x10" ::: "memory");
        else             asm volatile("s_wait_asynccnt 0x0"  ::: "memory");
        __syncthreads();

        // --- ctx += P . V over 16 cv tiles, V frags from LDS ---
        const u16* vc = vbuf[cur] + cl * 32 + h * 16;
        for (int t = 0; t < 16; ++t) {
            const u16* p = vc + t * 16 * 32;
            u16x8 v0 = *(const u16x8*)p;
            u16x8 v1 = *(const u16x8*)(p + 8);
            acc[t] = wmma_bf16(pf, mkfrag(v0, v1), acc[t]);
        }

        // all consumers done before anyone re-stages this buffer
        __syncthreads();
    }

    // --- normalize, write ctxT bf16 [B][N][Cv] ---
    float inv[8];
    for (int r = 0; r < 8; ++r) inv[r] = 1.0f / lrun[r];
    for (int t = 0; t < 16; ++t)
        for (int r = 0; r < 8; ++r) {
            int n = i0 + r + 8 * h;
            ctxT[((size_t)b * NN + n) * CV + t * 16 + cl] = f2bf(acc[t][r] * inv[r]);
        }
}

// ---------------------------------------------------------------------------
// Host-side launcher
// ---------------------------------------------------------------------------
extern "C" void kernel_launch(void* const* d_in, const int* in_sizes, int n_in,
                              void* d_out, int out_size, void* d_ws, size_t ws_size,
                              hipStream_t stream) {
    const float* x     = (const float*)d_in[0];
    const float* Wk    = (const float*)d_in[1];
    const float* bk    = (const float*)d_in[2];
    const float* gamma = (const float*)d_in[3];
    const float* beta  = (const float*)d_in[4];
    const float* Wv    = (const float*)d_in[5];
    const float* bv    = (const float*)d_in[6];
    const float* Ww    = (const float*)d_in[7];
    const float* bw    = (const float*)d_in[8];
    float* out = (float*)d_out;

    // d_out slots (floats)
    float* out_proj = out;                                    // [8,512,4096]
    float* out_f1   = out + (size_t)BB * CO * NN;             // [8,256,4096]
    float* out_f2   = out_f1 + (size_t)BB * CK * NN;
    float* out_val  = out_f2 + (size_t)BB * CK * NN;          // [8,256,4096]

    // workspace layout (bytes, 256-aligned by construction)
    char* ws = (char*)d_ws;
    u16*   xt_b   = (u16*)(ws);                               // [8][4096][512] bf16  33.5MB
    u16*   wk_b   = (u16*)(ws + 33554432);                    // 256KB
    u16*   wv_b   = (u16*)(ws + 33816576);
    u16*   ww_b   = (u16*)(ws + 34078720);
    float* qk_f   = (float*)(ws + 34340864);                  // [8][256][4096] f32   33.5MB
    u16*   feat_t = (u16*)(ws + 67895296);                    // [8][4096][256] bf16  16.8MB
    u16*   val_b  = (u16*)(ws + 84672512);                    // [8][256][4096] bf16  16.8MB
    u16*   ctx_t  = (u16*)(ws + 101449728);                   // [8][4096][256] bf16  16.8MB
    float* scsh   = (float*)(ws + 118226944);                 // 2*256 f32

    // 1) weights -> bf16
    k_cvt_bf16<<<(CK * CC + 255) / 256, 256, 0, stream>>>(Wk, wk_b, CK * CC);
    k_cvt_bf16<<<(CV * CC + 255) / 256, 256, 0, stream>>>(Wv, wv_b, CV * CC);
    k_cvt_bf16<<<(CO * CV + 255) / 256, 256, 0, stream>>>(Ww, ww_b, CO * CV);

    // 2) x -> bf16 transposed [B][N][C]
    k_transpose_cvt<<<dim3(NN / 32, CC / 32, BB), dim3(32, 8), 0, stream>>>(x, xt_b);

    // 3) qk = Wk*x + bk (f32 to ws);  4) value = Wv*x + bv (f32 to d_out, bf16 to ws)
    k_gemm<<<dim3(NN / 64, CK / 16, BB), 32, 0, stream>>>(wk_b, xt_b, bk, qk_f, nullptr, CK, CC, NN);
    k_gemm<<<dim3(NN / 64, CV / 16, BB), 32, 0, stream>>>(wv_b, xt_b, bv, out_val, val_b, CV, CC, NN);

    // 5) BN stats -> folded scale/shift
    k_bnstats<<<CK, 256, 0, stream>>>(qk_f, gamma, beta, scsh);

    // 6) feat = relu(BN(qk)): f32 to both tuple slots + bf16 transposed
    k_feat<<<dim3(NN / 32, CK / 32, BB), dim3(32, 8), 0, stream>>>(qk_f, scsh, out_f1, out_f2, feat_t);

    // 7) fused flash attention -> ctx_t bf16 [B][N][Cv]
    k_flash<<<dim3(NN / 16 / 2, BB), 64, 0, stream>>>(feat_t, val_b, ctx_t);

    // 8) out = Ww*ctx + bw (f32 to d_out)
    k_gemm<<<dim3(NN / 64, CO / 16, BB), 32, 0, stream>>>(ww_b, ctx_t, bw, out_proj, nullptr, CO, CV, NN);
}
